// RBFPseudoDerivativeLayer_36558761624239
// MI455X (gfx1250) — matrix operand order repaired and checked by reading, not compiled, not run on hardware
//
#include <hip/hip_runtime.h>
#include <math.h>

// Problem constants from the reference.
#define B_DIM   1024
#define OUT_DIM 512
#define IN_DIM  1024

#define MTILES  (B_DIM / 16)    // 64
#define NTILES  (OUT_DIM / 16)  // 32
#define KCHUNKS 64              // unified K = 2048 (phase0: x^2*u^2, phase1: x*(-2u^2w)), 32 per chunk

typedef __attribute__((ext_vector_type(16))) _Float16 v16h;
typedef __attribute__((ext_vector_type(8)))  float    v8f;

// ---------------------------------------------------------------------------
// Prep: C[o] = sum_i (u[o,i]*w[o,i])^2   (512 floats at d_ws offset 0)
// ---------------------------------------------------------------------------
__global__ __launch_bounds__(256) void rbf_prep_c(const float* __restrict__ u,
                                                  const float* __restrict__ w,
                                                  float* __restrict__ cvec) {
  __shared__ float red[256];
  const int o = blockIdx.x;
  const float* ur = u + (size_t)o * IN_DIM;
  const float* wr = w + (size_t)o * IN_DIM;
  float s = 0.f;
  for (int i = threadIdx.x; i < IN_DIM; i += 256) {
    float t = ur[i] * wr[i];
    s += t * t;
  }
  red[threadIdx.x] = s;
  __syncthreads();
  for (int k = 128; k > 0; k >>= 1) {
    if ((int)threadIdx.x < k) red[threadIdx.x] += red[threadIdx.x + k];
    __syncthreads();
  }
  if (threadIdx.x == 0) cvec[o] = red[0];
}

// ---------------------------------------------------------------------------
// Stage A fragments (f16, pre-swizzled to per-lane WMMA A layout, contiguous K order):
//   kc in [0,32):  halves = (x)^2 for K = (kc)*32 ...      (phase 0)
//   kc in [32,64): halves = (x)   for K = (kc-32)*32 ...   (phase 1)
// Fragment element j of lane l covers K = kb + j,
// kb = (l>>4)*16, row m = l&15.  One wave per (tm, kc).
// ---------------------------------------------------------------------------
__global__ __launch_bounds__(256) void rbf_stage_a(const float* __restrict__ X,
                                                   _Float16* __restrict__ Afeat) {
  const int gw   = (int)((blockIdx.x * 256 + threadIdx.x) >> 5); // 0..4095
  const int lane = threadIdx.x & 31;
  const int tm    = gw >> 6;          // 0..63
  const int kc    = gw & 63;          // 0..63
  const int phase = kc >> 5;          // 0: x^2, 1: x
  const int k0    = (kc & 31) * 32;
  const int m  = lane & 15;
  const int kb = (lane >> 4) * 16;

  const float* p = X + (size_t)(tm * 16 + m) * IN_DIM + k0 + kb;
  float fa[16];
  *(float4*)(fa + 0)  = *(const float4*)(p + 0);
  *(float4*)(fa + 4)  = *(const float4*)(p + 4);
  *(float4*)(fa + 8)  = *(const float4*)(p + 8);
  *(float4*)(fa + 12) = *(const float4*)(p + 12);

  v16h f;
#pragma unroll
  for (int j = 0; j < 16; ++j) {
    float v = fa[j];
    f[j] = phase ? (_Float16)v : (_Float16)(v * v);
  }
  *(v16h*)(Afeat + ((size_t)(tm * KCHUNKS + kc) * 32 + lane) * 16) = f;
}

// ---------------------------------------------------------------------------
// Stage B fragments (f16, per-lane WMMA B layout, contiguous K order as in A):
//   phase 0: u^2 ; phase 1: -2*u^2*w
// Fragment element j of lane l covers K = (l>>4)*16 + j, col n = l&15.
// One wave per (tn, kc).
// ---------------------------------------------------------------------------
__global__ __launch_bounds__(256) void rbf_stage_b(const float* __restrict__ U,
                                                   const float* __restrict__ W,
                                                   _Float16* __restrict__ Bfeat) {
  const int gw   = (int)((blockIdx.x * 256 + threadIdx.x) >> 5); // 0..2047
  const int lane = threadIdx.x & 31;
  const int tn    = gw >> 6;          // 0..31
  const int kc    = gw & 63;          // 0..63
  const int phase = kc >> 5;          // 0: u^2, 1: -2u^2w
  const int k0    = (kc & 31) * 32;
  const int n  = lane & 15;
  const int kb = (lane >> 4) * 16;

  const float* up = U + (size_t)(tn * 16 + n) * IN_DIM + k0 + kb;
  const float* wp = W + (size_t)(tn * 16 + n) * IN_DIM + k0 + kb;
  float fu[16], fw[16];
  *(float4*)(fu + 0)  = *(const float4*)(up + 0);
  *(float4*)(fu + 4)  = *(const float4*)(up + 4);
  *(float4*)(fu + 8)  = *(const float4*)(up + 8);
  *(float4*)(fu + 12) = *(const float4*)(up + 12);
  *(float4*)(fw + 0)  = *(const float4*)(wp + 0);
  *(float4*)(fw + 4)  = *(const float4*)(wp + 4);
  *(float4*)(fw + 8)  = *(const float4*)(wp + 8);
  *(float4*)(fw + 12) = *(const float4*)(wp + 12);

  v16h f;
#pragma unroll
  for (int j = 0; j < 16; ++j) {
    float uu = fu[j] * fu[j];
    f[j] = phase ? (_Float16)(-2.0f * uu * fw[j]) : (_Float16)uu;
  }
  *(v16h*)(Bfeat + ((size_t)(tn * KCHUNKS + kc) * 32 + lane) * 16) = f;
}

// ---------------------------------------------------------------------------
// Shared epilogue: z = acc + C[o]; y = exp(-z); out = y + a01*(1-2y)
// C/D layout: VGPR r -> M = r + 8*(lane>>4);  N = lane&15.
// ---------------------------------------------------------------------------
__device__ __forceinline__ void rbf_epilogue(const v8f acc[2][2], int rowBase,
                                             int colBase, int ln16, int hi,
                                             const float* __restrict__ cvec,
                                             const float* __restrict__ A01,
                                             float* __restrict__ out) {
#pragma unroll
  for (int nt = 0; nt < 2; ++nt) {
    const int col  = colBase + nt * 16 + ln16;
    const float c  = cvec[col];
    const float a0 = A01[col];
#pragma unroll
    for (int mt = 0; mt < 2; ++mt) {
      const int row0 = rowBase + mt * 16 + hi * 8;
#pragma unroll
      for (int r = 0; r < 8; ++r) {
        float z = acc[mt][nt][r] + c;
        float y = __expf(-z);
        out[(size_t)(row0 + r) * OUT_DIM + col] = fmaf(a0, 1.0f - 2.0f * y, y);
      }
    }
  }
}

// ---------------------------------------------------------------------------
// Main GEMM on pre-staged f16 fragments. Wave tile 32x32 (2x2 WMMA accums),
// block = 8 wave32s as 2(M) x 4(N) -> 64x128 tile. Grid (4,16).
// Inner loop: 4 contiguous 32B v16h loads + 4 WMMAs.
// ---------------------------------------------------------------------------
__global__ __launch_bounds__(256) void rbf_wmma_gemm_staged(
    const _Float16* __restrict__ Afeat, const _Float16* __restrict__ Bfeat,
    const float* __restrict__ A01, const float* __restrict__ cvec,
    float* __restrict__ out) {
  const int lane   = threadIdx.x & 31;
  const int waveId = threadIdx.x >> 5;
  const int waveM  = waveId >> 2;   // 0..1
  const int waveN  = waveId & 3;    // 0..3
  const int tmBase = blockIdx.y * 4 + waveM * 2;   // first of 2 M tiles
  const int tnBase = blockIdx.x * 8 + waveN * 2;   // first of 2 N tiles

  const _Float16* ap0 = Afeat + ((size_t)(tmBase + 0) * KCHUNKS * 32 + lane) * 16;
  const _Float16* ap1 = Afeat + ((size_t)(tmBase + 1) * KCHUNKS * 32 + lane) * 16;
  const _Float16* bp0 = Bfeat + ((size_t)(tnBase + 0) * KCHUNKS * 32 + lane) * 16;
  const _Float16* bp1 = Bfeat + ((size_t)(tnBase + 1) * KCHUNKS * 32 + lane) * 16;

  v8f acc[2][2] = {};

#pragma unroll 4
  for (int kc = 0; kc < KCHUNKS; ++kc) {
    const size_t off = (size_t)kc * 32 * 16;   // halves per k-chunk per tile
    v16h a0 = *(const v16h*)(ap0 + off);
    v16h a1 = *(const v16h*)(ap1 + off);
    v16h b0 = *(const v16h*)(bp0 + off);
    v16h b1 = *(const v16h*)(bp1 + off);
    acc[0][0] = __builtin_amdgcn_wmma_f32_16x16x32_f16(false, a0, false, b0,
                                                       (short)0, acc[0][0], false, false);
    acc[0][1] = __builtin_amdgcn_wmma_f32_16x16x32_f16(false, a0, false, b1,
                                                       (short)0, acc[0][1], false, false);
    acc[1][0] = __builtin_amdgcn_wmma_f32_16x16x32_f16(false, a1, false, b0,
                                                       (short)0, acc[1][0], false, false);
    acc[1][1] = __builtin_amdgcn_wmma_f32_16x16x32_f16(false, a1, false, b1,
                                                       (short)0, acc[1][1], false, false);
  }

  rbf_epilogue(acc, tmBase * 16, tnBase * 16, lane & 15, lane >> 4, cvec, A01, out);
}

// ---------------------------------------------------------------------------
// Fallback (small d_ws): fused conversion GEMM from round 1.
// ---------------------------------------------------------------------------
__global__ __launch_bounds__(256) void rbf_wmma_gemm_fused(
    const float* __restrict__ X, const float* __restrict__ U,
    const float* __restrict__ W, const float* __restrict__ A01,
    const float* __restrict__ cvec, float* __restrict__ out) {
  const int lane   = threadIdx.x & 31;
  const int waveId = threadIdx.x >> 5;
  const int waveM  = waveId >> 2;
  const int waveN  = waveId & 3;
  const int rowBase = blockIdx.y * 64 + waveM * 32;
  const int colBase = blockIdx.x * 128 + waveN * 32;

  const int ln16 = lane & 15;
  const int hi   = lane >> 4;
  const int kbA  = hi * 16;
  const int kbB  = hi * 16;

  const float* axp[2];
  const float* pup[2];
  const float* pwp[2];
#pragma unroll
  for (int t = 0; t < 2; ++t) {
    axp[t] = X + (size_t)(rowBase + t * 16 + ln16) * IN_DIM + kbA;
    pup[t] = U + (size_t)(colBase + t * 16 + ln16) * IN_DIM + kbB;
    pwp[t] = W + (size_t)(colBase + t * 16 + ln16) * IN_DIM + kbB;
  }

  v8f acc[2][2] = {};

  for (int k0 = 0; k0 < IN_DIM; k0 += 32) {
    v16h fx[2], fx2[2];
#pragma unroll
    for (int mt = 0; mt < 2; ++mt) {
      float fa[16];
      const float* p = axp[mt] + k0;
      *(float4*)(fa + 0)  = *(const float4*)(p + 0);
      *(float4*)(fa + 4)  = *(const float4*)(p + 4);
      *(float4*)(fa + 8)  = *(const float4*)(p + 8);
      *(float4*)(fa + 12) = *(const float4*)(p + 12);
#pragma unroll
      for (int j = 0; j < 16; ++j) {
        float v = fa[j];
        fx[mt][j]  = (_Float16)v;
        fx2[mt][j] = (_Float16)(v * v);
      }
    }
    v16h fu2[2], fg[2];
#pragma unroll
    for (int nt = 0; nt < 2; ++nt) {
      float fu[16], fw[16];
      const float* up = pup[nt] + k0;
      const float* wp = pwp[nt] + k0;
      *(float4*)(fu + 0)  = *(const float4*)(up + 0);
      *(float4*)(fu + 4)  = *(const float4*)(up + 4);
      *(float4*)(fu + 8)  = *(const float4*)(up + 8);
      *(float4*)(fu + 12) = *(const float4*)(up + 12);
      *(float4*)(fw + 0)  = *(const float4*)(wp + 0);
      *(float4*)(fw + 4)  = *(const float4*)(wp + 4);
      *(float4*)(fw + 8)  = *(const float4*)(wp + 8);
      *(float4*)(fw + 12) = *(const float4*)(wp + 12);
#pragma unroll
      for (int j = 0; j < 16; ++j) {
        float uu = fu[j] * fu[j];
        fu2[nt][j] = (_Float16)uu;
        fg[nt][j]  = (_Float16)(-2.0f * uu * fw[j]);
      }
    }
#pragma unroll
    for (int mt = 0; mt < 2; ++mt) {
#pragma unroll
      for (int nt = 0; nt < 2; ++nt) {
        acc[mt][nt] = __builtin_amdgcn_wmma_f32_16x16x32_f16(
            false, fx2[mt], false, fu2[nt], (short)0, acc[mt][nt], false, false);
        acc[mt][nt] = __builtin_amdgcn_wmma_f32_16x16x32_f16(
            false, fx[mt], false, fg[nt], (short)0, acc[mt][nt], false, false);
      }
    }
  }

  rbf_epilogue(acc, rowBase, colBase, ln16, hi, cvec, A01, out);
}

// ---------------------------------------------------------------------------
extern "C" void kernel_launch(void* const* d_in, const int* in_sizes, int n_in,
                              void* d_out, int out_size, void* d_ws, size_t ws_size,
                              hipStream_t stream) {
  const float* x   = (const float*)d_in[0];  // [B, IN]
  const float* w   = (const float*)d_in[1];  // [OUT, IN]
  const float* u   = (const float*)d_in[2];  // [OUT, IN]
  const float* a01 = (const float*)d_in[3];  // [1, OUT]
  float* out = (float*)d_out;                // [B, OUT]

  // Workspace layout: [cvec: 2KB][Afeat: 4MB][Bfeat: 2MB]
  const size_t A_OFF   = 2048;
  const size_t A_BYTES = (size_t)MTILES * KCHUNKS * 32 * 16 * sizeof(_Float16); // 4 MB
  const size_t B_OFF   = A_OFF + A_BYTES;
  const size_t B_BYTES = (size_t)NTILES * KCHUNKS * 32 * 16 * sizeof(_Float16); // 2 MB
  const size_t NEED    = B_OFF + B_BYTES;

  float* cvec = (float*)d_ws;
  rbf_prep_c<<<OUT_DIM, 256, 0, stream>>>(u, w, cvec);

  dim3 grid(OUT_DIM / 128, B_DIM / 64);
  if (ws_size >= NEED) {
    _Float16* Afeat = (_Float16*)((char*)d_ws + A_OFF);
    _Float16* Bfeat = (_Float16*)((char*)d_ws + B_OFF);
    // 4096 A-staging waves, 2048 B-staging waves (8 waves per block).
    rbf_stage_a<<<(MTILES * KCHUNKS) / 8, 256, 0, stream>>>(x, Afeat);
    rbf_stage_b<<<(NTILES * KCHUNKS) / 8, 256, 0, stream>>>(u, w, Bfeat);
    rbf_wmma_gemm_staged<<<grid, 256, 0, stream>>>(Afeat, Bfeat, a01, cvec, out);
  } else {
    rbf_wmma_gemm_fused<<<grid, 256, 0, stream>>>(x, u, w, a01, cvec, out);
  }
}
